// Decoder_81406810129191
// MI455X (gfx1250) — matrix operand order repaired and checked by reading, not compiled, not run on hardware
//
#include <hip/hip_runtime.h>

// Problem dims (fixed by reference)
#define B_ 16
#define T_ 128
#define V_ 32000
#define E_ 512
#define H_ 1024
#define G_ 4096  // 4*H

#define TKS 128        // K elements per LDS slab (4 WMMA k-steps)
#define LDSS 136       // padded LDS row stride in elements (272 B, 16B-aligned)

typedef __attribute__((ext_vector_type(8)))  __bf16 v8bf;
typedef __attribute__((ext_vector_type(16))) __bf16 v16bf;
typedef __attribute__((ext_vector_type(8)))  float  v8f;

__device__ __forceinline__ v16bf cat16(v8bf lo, v8bf hi) {
  return __builtin_shufflevector(lo, hi, 0,1,2,3,4,5,6,7,8,9,10,11,12,13,14,15);
}

__device__ __forceinline__ float sigmoidf_(float x) { return 1.f / (1.f + __expf(-x)); }

// Async global -> LDS copy of one 16-byte chunk (tracked by ASYNCcnt)
__device__ __forceinline__ void async_b128(unsigned lds_off, const void* gptr) {
  asm volatile("global_load_async_to_lds_b128 %0, %1, off"
               :: "v"(lds_off), "v"((unsigned long long)(uintptr_t)gptr)
               : "memory");
}

// Copy a 64-row x 128-element (bf16) B slab into LDS.
// 1024 16B-chunks, 8 per thread, consecutive threads take consecutive chunks.
// Each wave issues exactly 8 async instructions per slab.
__device__ __forceinline__ void copy_slab_async(
    __bf16* lds, const __bf16* gsrc /* = Bt + n0*K + k0 */, int K, int tid)
{
#pragma unroll
  for (int i = 0; i < 8; ++i) {
    const int c   = i * 128 + tid;
    const int row = c >> 4;     // 16 chunks per row
    const int sub = c & 15;
    const unsigned lds_off =
        (unsigned)(uintptr_t)(lds + row * LDSS + sub * 8);  // low 32 bits = LDS offset
    async_b128(lds_off, gsrc + (size_t)row * K + sub * 8);
  }
}

// 4 WMMA k-steps over one LDS slab (K = 128). Scheduler hints group the
// loads ahead of the 4 back-to-back WMMAs for each k-step.
__device__ __forceinline__ void gemm_slab(
    const __bf16* __restrict__ arow, const __bf16* lb,
    int ln, int half, v8f acc[4])
{
#pragma unroll
  for (int ks = 0; ks < 4; ++ks) {
    const int k0 = ks * 32;
    v16bf a = cat16(*(const v8bf*)(arow + k0), *(const v8bf*)(arow + k0 + 16));
    v16bf bb[4];
#pragma unroll
    for (int j = 0; j < 4; ++j) {
      const __bf16* bp = lb + (j * 16 + ln) * LDSS + k0 + half * 8;
      bb[j] = cat16(*(const v8bf*)bp, *(const v8bf*)(bp + 16));
    }
#pragma unroll
    for (int j = 0; j < 4; ++j)
      acc[j] = __builtin_amdgcn_wmma_f32_16x16x32_bf16(
          false, a, false, bb[j], (short)0, acc[j], false, false);
    __builtin_amdgcn_sched_group_barrier(0x020, 2, 0);  // 2 VMEM reads (A frag)
    __builtin_amdgcn_sched_group_barrier(0x100, 8, 0);  // 8 DS reads (B frags)
    __builtin_amdgcn_sched_group_barrier(0x008, 4, 0);  // 4 WMMA
  }
}

// ---------------------------------------------------------------------------
// Tiled transpose + fp32 -> bf16 convert: src [K,N] row-major -> dst [N,K]
// ---------------------------------------------------------------------------
__global__ void __launch_bounds__(256) transpose_to_bf16(
    const float* __restrict__ src, __bf16* __restrict__ dst, int K, int N)
{
  __shared__ float tile[32][33];
  const int nb = blockIdx.x * 32;
  const int kb = blockIdx.y * 32;
  const int tx = threadIdx.x;   // 0..31
  const int ty = threadIdx.y;   // 0..7
#pragma unroll
  for (int i = 0; i < 32; i += 8)
    tile[ty + i][tx] = src[(size_t)(kb + ty + i) * N + (nb + tx)];
  __syncthreads();
#pragma unroll
  for (int i = 0; i < 32; i += 8)
    dst[(size_t)(nb + ty + i) * K + (kb + tx)] = (__bf16)tile[tx][ty + i];
}

// ---------------------------------------------------------------------------
// Embedding gather -> bf16 A matrix, row r = t*B + b (for zx in [T,B,4H] order)
// ---------------------------------------------------------------------------
__global__ void __launch_bounds__(128) embed_gather(
    const int* __restrict__ tokens, const float* __restrict__ emb,
    __bf16* __restrict__ Abuf)
{
  const int r = blockIdx.x;   // r = t*16 + b
  const int t = r >> 4;
  const int b = r & 15;
  const int tok = tokens[b * T_ + t];
  const float* src = emb + (size_t)tok * E_;
  __bf16* dst = Abuf + (size_t)r * E_;
  for (int e = threadIdx.x; e < E_; e += 128) dst[e] = (__bf16)src[e];
}

__global__ void __launch_bounds__(256) init_state(
    const float* __restrict__ h0, const float* __restrict__ c0,
    __bf16* __restrict__ hb, float* __restrict__ cb)
{
  const int i = blockIdx.x * 256 + threadIdx.x;
  if (i < B_ * H_) { hb[i] = (__bf16)h0[i]; cb[i] = c0[i]; }
}

// ---------------------------------------------------------------------------
// Block-tiled WMMA bf16 GEMM:  C[M,N] = A[M,K] * Bt[N,K]^T + bias[N]
// Block = 4 waves computing 64x64. B slab (64 x 128) double-buffered in LDS
// via global_load_async_to_lds_b128 (8 async instrs/wave/slab); each wave
// owns a 16-row slice of A loaded directly from global.
// ---------------------------------------------------------------------------
__global__ void __launch_bounds__(128) wmma_gemm_bf16(
    const __bf16* __restrict__ A, const __bf16* __restrict__ Bt,
    float* __restrict__ C, const float* __restrict__ bias,
    int M, int N, int K)
{
  __shared__ __bf16 bt[2][64 * LDSS];   // 2 x 17 KB

  const int tid  = threadIdx.x;
  const int wave = tid >> 5;
  const int l    = tid & 31;
  const int half = l >> 4;
  const int ln   = l & 15;
  const int m0   = blockIdx.y * 64 + wave * 16;
  const int n0   = blockIdx.x * 64;

  v8f acc[4];
#pragma unroll
  for (int j = 0; j < 4; ++j)
#pragma unroll
    for (int i = 0; i < 8; ++i) acc[j][i] = 0.f;

  const __bf16* arow = A  + (size_t)(m0 + ln) * K + half * 8;
  const __bf16* gB   = Bt + (size_t)n0 * K;

  const int nk = K / TKS;               // >= 4 for all our shapes
  copy_slab_async(&bt[0][0], gB, K, tid);

  for (int kk = 0; kk < nk - 1; ++kk) {
    copy_slab_async(&bt[(kk + 1) & 1][0], gB + (size_t)(kk + 1) * TKS, K, tid);
    asm volatile("s_wait_asynccnt 0x8" ::: "memory");  // current slab landed
    __syncthreads();
    gemm_slab(arow + (size_t)kk * TKS, &bt[kk & 1][0], ln, half, acc);
    __syncthreads();
  }
  asm volatile("s_wait_asynccnt 0x0" ::: "memory");
  __syncthreads();
  gemm_slab(arow + (size_t)(nk - 1) * TKS, &bt[(nk - 1) & 1][0], ln, half, acc);

#pragma unroll
  for (int j = 0; j < 4; ++j) {
    const int col = n0 + j * 16 + ln;
    const float bv = bias[col];
#pragma unroll
    for (int v = 0; v < 8; ++v) {
      const int row = m0 + half * 8 + v;
      C[(size_t)row * N + col] = acc[j][v] + bv;
    }
  }
}

// ---------------------------------------------------------------------------
// One LSTM timestep. Block = 4 waves, one 16-hidden-unit tile j0..j0+15.
// K = 1024 is split across the 4 waves (256 each -> 8 dependent WMMA steps,
// quartering the serial chain), partials reduced through LDS, wave 0 gates.
// ---------------------------------------------------------------------------
__global__ void __launch_bounds__(128) lstm_step(
    const __bf16* __restrict__ hcur, const __bf16* __restrict__ Wht,
    const float* __restrict__ zx, float* __restrict__ cbuf,
    __bf16* __restrict__ hnext, __bf16* __restrict__ hs, int t)
{
  __shared__ float red[4 * 4 * 32 * 8];   // [wave][gate][lane][v] = 16 KB

  const int tid  = threadIdx.x;
  const int wave = tid >> 5;
  const int l    = tid & 31;
  const int half = l >> 4;
  const int ln   = l & 15;
  const int j0   = blockIdx.x * 16;

  v8f acc[4];
#pragma unroll
  for (int j = 0; j < 4; ++j)
#pragma unroll
    for (int i = 0; i < 8; ++i) acc[j][i] = 0.f;

  const int kbeg = wave * 256;            // this wave's K slice
  const __bf16* arow = hcur + (size_t)ln * H_ + kbeg + half * 8;
  const __bf16* brow = Wht + (size_t)(j0 + ln) * H_ + kbeg + half * 8;

  for (int k0 = 0; k0 < 256; k0 += 32) {
    v16bf a = cat16(*(const v8bf*)(arow + k0), *(const v8bf*)(arow + k0 + 16));
    v16bf bb[4];
#pragma unroll
    for (int gi = 0; gi < 4; ++gi) {
      const __bf16* bp = brow + ((size_t)gi * H_ * H_) + k0;  // gate gi: rows gi*H + j
      bb[gi] = cat16(*(const v8bf*)bp, *(const v8bf*)(bp + 16));
    }
#pragma unroll
    for (int gi = 0; gi < 4; ++gi)
      acc[gi] = __builtin_amdgcn_wmma_f32_16x16x32_bf16(
          false, a, false, bb[gi], (short)0, acc[gi], false, false);
  }

  // Cross-wave K reduction through LDS
#pragma unroll
  for (int gi = 0; gi < 4; ++gi)
    *(v8f*)&red[((wave * 4 + gi) * 32 + l) * 8] = acc[gi];
  __syncthreads();

  if (wave == 0) {
#pragma unroll
    for (int gi = 0; gi < 4; ++gi) {
      v8f s = *(const v8f*)&red[(gi * 32 + l) * 8];
#pragma unroll
      for (int w = 1; w < 4; ++w) {
        v8f p = *(const v8f*)&red[((w * 4 + gi) * 32 + l) * 8];
#pragma unroll
        for (int i = 0; i < 8; ++i) s[i] += p[i];
      }
      acc[gi] = s;
    }

    const int j = j0 + ln;
#pragma unroll
    for (int v = 0; v < 8; ++v) {
      const int b = half * 8 + v;
      const float* zr = zx + ((size_t)t * B_ + b) * G_;
      const float zi = acc[0][v] + zr[j];
      const float zf = acc[1][v] + zr[H_ + j];
      const float zg = acc[2][v] + zr[2 * H_ + j];
      const float zo = acc[3][v] + zr[3 * H_ + j];
      const float cn = sigmoidf_(zf) * cbuf[b * H_ + j] + sigmoidf_(zi) * tanhf(zg);
      const float hn = sigmoidf_(zo) * tanhf(cn);
      cbuf[b * H_ + j] = cn;
      hnext[b * H_ + j] = (__bf16)hn;
      hs[((size_t)b * T_ + t) * H_ + j] = (__bf16)hn;   // row b*T+t -> [B,T,V] output
    }
  }
}

// ---------------------------------------------------------------------------
extern "C" void kernel_launch(void* const* d_in, const int* in_sizes, int n_in,
                              void* d_out, int out_size, void* d_ws, size_t ws_size,
                              hipStream_t stream)
{
  (void)in_sizes; (void)n_in; (void)out_size; (void)ws_size;

  const int*   tokens = (const int*)  d_in[0];
  const float* h0     = (const float*)d_in[1];
  const float* c0     = (const float*)d_in[2];
  const float* emb    = (const float*)d_in[3];
  const float* Wx     = (const float*)d_in[4];
  const float* Wh     = (const float*)d_in[5];
  const float* bvec   = (const float*)d_in[6];
  const float* fcW    = (const float*)d_in[7];
  const float* fcb    = (const float*)d_in[8];
  float* out = (float*)d_out;

  // Workspace carve-out (~116 MB total)
  char* base = (char*)d_ws;
  size_t off = 0;
  auto alloc = [&](size_t bytes) -> void* {
    off = (off + 255) & ~(size_t)255;
    void* p = base + off;
    off += bytes;
    return p;
  };

  __bf16* Wxt  = (__bf16*)alloc((size_t)G_ * E_ * 2);        //  4 MB  [4096, 512]
  __bf16* Wht  = (__bf16*)alloc((size_t)G_ * H_ * 2);        //  8 MB  [4096,1024]
  __bf16* fcWt = (__bf16*)alloc((size_t)V_ * H_ * 2);        // 62 MB  [32000,1024]
  __bf16* Abuf = (__bf16*)alloc((size_t)B_ * T_ * E_ * 2);   //  2 MB
  float*  zx   = (float*) alloc((size_t)B_ * T_ * G_ * 4);   // 32 MB  [T,B,4H]
  __bf16* hs   = (__bf16*)alloc((size_t)B_ * T_ * H_ * 2);   //  4 MB  rows b*T+t
  __bf16* hb0  = (__bf16*)alloc((size_t)B_ * H_ * 2);
  __bf16* hb1  = (__bf16*)alloc((size_t)B_ * H_ * 2);
  float*  cbuf = (float*) alloc((size_t)B_ * H_ * 4);

  dim3 tb(32, 8);
  // Wx [E,4H] -> Wxt [4H,E] ; Wh [H,4H] -> Wht [4H,H] ; fcW [H,V] -> fcWt [V,H]
  transpose_to_bf16<<<dim3(G_/32, E_/32), tb, 0, stream>>>(Wx,  Wxt,  E_, G_);
  transpose_to_bf16<<<dim3(G_/32, H_/32), tb, 0, stream>>>(Wh,  Wht,  H_, G_);
  transpose_to_bf16<<<dim3(V_/32, H_/32), tb, 0, stream>>>(fcW, fcWt, H_, V_);

  embed_gather<<<B_*T_, 128, 0, stream>>>(tokens, emb, Abuf);
  init_state<<<(B_*H_ + 255)/256, 256, 0, stream>>>(h0, c0, hb0, cbuf);

  // zx[T,B,4H] = Abuf @ Wx + b   (M=2048, N=4096, K=512)  — 64x64 block tiles
  wmma_gemm_bf16<<<dim3(G_/64, (B_*T_)/64), 128, 0, stream>>>(
      Abuf, Wxt, zx, bvec, B_*T_, G_, E_);

  // Recurrence: stream ordering enforces the step dependency
  __bf16* hA = hb0; __bf16* hB = hb1;
  for (int t = 0; t < T_; ++t) {
    lstm_step<<<H_/16, 128, 0, stream>>>(hA, Wht, zx, cbuf, hB, hs, t);
    __bf16* tmp = hA; hA = hB; hB = tmp;
  }

  // logits[B,T,V] = hs @ fcW + fcb   (M=2048, N=32000, K=1024)
  wmma_gemm_bf16<<<dim3(V_/64, (B_*T_)/64), 128, 0, stream>>>(
      hs, fcWt, out, fcb, B_*T_, V_, H_);
}
